// GDSR_14688788152895
// MI455X (gfx1250) — matrix operand order repaired and checked by reading, not compiled, not run on hardware
//
#include <hip/hip_runtime.h>
#include <hip/hip_bf16.h>
#include <math.h>

typedef __attribute__((ext_vector_type(16))) _Float16 v16h;
typedef __attribute__((ext_vector_type(8)))  float    v8f;

constexpr int IMG   = 256;
constexpr int BATCH = 16;
constexpr size_t HW = (size_t)IMG * IMG;   // 65536

// ---------------------------------------------------------------------------
// Layer-1 convs (Cin<=4 -> 32, 3x3, SAME, ReLU), fp32 NCHW in -> f16 NHWC out
// grid (IMG /*y*/, BATCH /*b*/), block 256 (x)
// ---------------------------------------------------------------------------
__global__ __launch_bounds__(256) void conv_l1(
    const float* __restrict__ in1, int c1,
    const float* __restrict__ in2, int c2,
    const float* __restrict__ w, const float* __restrict__ bias,
    _Float16* __restrict__ out)            // [b][y][x][32]
{
  const int x = threadIdx.x;
  const int y = blockIdx.x;
  const int b = blockIdx.y;
  const int cin = c1 + c2;
  __shared__ float sw[32 * 4 * 9 + 32];
  for (int i = threadIdx.x; i < 32 * cin * 9; i += 256) sw[i] = w[i];
  float* sb = sw + 32 * cin * 9;
  for (int i = threadIdx.x; i < 32; i += 256) sb[i] = bias[i];
  __syncthreads();

  float inreg[4][9];
  for (int c = 0; c < cin; ++c) {
    const float* src = (c < c1) ? in1 + (size_t)(b * c1 + c) * HW
                                : in2 + (size_t)(b * c2 + (c - c1)) * HW;
    #pragma unroll
    for (int t = 0; t < 9; ++t) {
      int yy = y + t / 3 - 1, xx = x + t % 3 - 1;
      inreg[c][t] = (yy >= 0 && yy < IMG && xx >= 0 && xx < IMG)
                        ? src[(size_t)yy * IMG + xx] : 0.f;
    }
  }
  alignas(16) _Float16 ov[32];
  #pragma unroll
  for (int co = 0; co < 32; ++co) {
    float acc = sb[co];
    for (int c = 0; c < cin; ++c) {
      const float* wr = sw + (co * cin + c) * 9;
      #pragma unroll
      for (int t = 0; t < 9; ++t) acc = fmaf(inreg[c][t], wr[t], acc);
    }
    ov[co] = (_Float16)fmaxf(acc, 0.f);
  }
  uint4* d4 = (uint4*)(out + (((size_t)b * IMG + y) * IMG + x) * 32);
  const uint4* s4 = (const uint4*)ov;
  #pragma unroll
  for (int j = 0; j < 4; ++j) d4[j] = s4[j];   // 64B contiguous per pixel
}

// ---------------------------------------------------------------------------
// Pack (32,32,3,3) fp32 weights into WMMA B fragments (f16).
// wp layout: [ct(2)][tap(9)][lane(32)][e(16)]; element e of lane L:
// k(=cin) = 16*(L>>4) + e, n(=cout in tile) = L&15.
// ---------------------------------------------------------------------------
__global__ void pack_w(const float* __restrict__ w, _Float16* __restrict__ wp)
{
  int i = blockIdx.x * 256 + threadIdx.x;
  if (i >= 2 * 9 * 32 * 16) return;
  int e    = i & 15;
  int lane = (i >> 4) & 31;
  int t    = (i >> 9) % 9;
  int ct   = i / (9 * 32 * 16);
  int cout = ct * 16 + (lane & 15);
  int cin  = (lane >> 4) * 16 + e;
  wp[i] = (_Float16)w[((cout * 32 + cin) * 3 + t / 3) * 3 + t % 3];
}

// ---------------------------------------------------------------------------
// 32->32 3x3 SAME conv, V_WMMA_F32_16X16X32_F16 implicit GEMM.
// Block = 256 threads (8 waves) per (row,batch); shared LDS tile [3][258][32].
// NHWC rows are contiguous 16KB spans -> staged with per-lane
// GLOBAL_LOAD_ASYNC_TO_LDS_B128 (ASYNCcnt), completed by s_wait_asynccnt.
// Wave w computes pixel tiles 2w and 2w+1: 9 taps x 2 cout tiles = 18 WMMAs each.
// ---------------------------------------------------------------------------
__global__ __launch_bounds__(256) void conv32_wmma(
    const _Float16* __restrict__ act,    // NHWC [b][y][x][32]
    const _Float16* __restrict__ wpack,  // [2][9][32][16]
    const float*    __restrict__ bias,   // [32]
    _Float16* __restrict__ out,          // NHWC [b][y][x][Ctot]
    int Ctot, int cout_off, int relu)
{
  const int tid  = threadIdx.x;
  const int lane = tid & 31;
  const int wid  = tid >> 5;
  const int y = blockIdx.x;
  const int b = blockIdx.y;
  __shared__ _Float16 tile[3 * 258 * 32];   // [r][col][cin], 49.5 KB

  // ---- stage rows y-1..y+1: async bulk copy for valid rows, ds-zero others
  for (int r = 0; r < 3; ++r) {
    int yy = y + r - 1;
    if (yy >= 0 && yy < IMG) {
      unsigned long long src =
          (unsigned long long)(size_t)(act + (((size_t)b * IMG + yy) * IMG) * 32);
      unsigned ldsbase = (unsigned)(size_t)(&tile[((size_t)r * 258 + 1) * 32]);
      for (int i = tid; i < 1024; i += 256) {        // 1024 x 16B = 16KB row
        unsigned lds = ldsbase + (unsigned)i * 16u;
        unsigned long long ga = src + (unsigned long long)i * 16ull;
        asm volatile("global_load_async_to_lds_b128 %0, %1, off"
                     :: "v"(lds), "v"(ga) : "memory");
      }
    } else {
      unsigned* t32 = (unsigned*)(tile + (size_t)r * 258 * 32);
      for (int i = tid; i < 258 * 32 / 2; i += 256) t32[i] = 0u;
    }
  }
  // ---- zero padding columns (col 0 and 257)
  for (int i = tid; i < 3 * 2 * 32; i += 256) {
    int r = i / 64, c = (i >> 5) & 1, cin = i & 31;
    tile[((size_t)r * 258 + (c ? 257 : 0)) * 32 + cin] = (_Float16)0.f;
  }
  asm volatile("s_wait_asynccnt 0x0" ::: "memory");
  __syncthreads();

  // ---- per-wave B fragments: 2 cout tiles x 9 taps
  v16h bw[2][9];
  #pragma unroll
  for (int ct = 0; ct < 2; ++ct)
    #pragma unroll
    for (int t = 0; t < 9; ++t)
      bw[ct][t] = ((const v16h*)wpack)[(ct * 9 + t) * 32 + lane];

  const int m  = lane & 15;   // A: M index / D: N index
  const int hh = lane >> 4;

  for (int xi = 0; xi < 2; ++xi) {
    const int x0 = (wid * 2 + xi) * 16;
    v8f acc0 = {}; v8f acc1 = {};
    #pragma unroll
    for (int t = 0; t < 9; ++t) {
      const int r   = t / 3;
      const int dx  = t % 3 - 1;
      const int col = x0 + m + dx + 1;
      const unsigned* lrow = (const unsigned*)(tile + ((size_t)r * 258 + col) * 32);
      union { v16h h; unsigned u[8]; } a;
      #pragma unroll
      for (int j = 0; j < 8; ++j) {            // A 16x32 f16 layout (pairs cin-contig)
        int cin0 = hh * 16 + j * 2;
        a.u[j] = lrow[cin0 >> 1];
      }
      acc0 = __builtin_amdgcn_wmma_f32_16x16x32_f16(false, a.h, false, bw[0][t],
                                                    (short)0, acc0, false, false);
      acc1 = __builtin_amdgcn_wmma_f32_16x16x32_f16(false, a.h, false, bw[1][t],
                                                    (short)0, acc1, false, false);
    }
    #pragma unroll
    for (int ct = 0; ct < 2; ++ct) {
      const v8f accv = ct ? acc1 : acc0;
      const int co   = cout_off + ct * 16 + m;   // D: N = lane&15
      const float bb = bias[ct * 16 + m];
      #pragma unroll
      for (int j = 0; j < 8; ++j) {              // D: M(pixel) = j + 8*hh
        const int x = x0 + j + hh * 8;
        float v = accv[j] + bb;
        if (relu) v = fmaxf(v, 0.f);
        out[(((size_t)b * IMG + y) * IMG + x) * Ctot + co] = (_Float16)v;
      }
    }
  }
}

// ---------------------------------------------------------------------------
// 32 -> 1 head conv (f16 NHWC in, fp32 out)
// ---------------------------------------------------------------------------
__global__ __launch_bounds__(256) void conv_vw3(
    const _Float16* __restrict__ v2,     // NHWC [b][y][x][32]
    const float* __restrict__ w, const float* __restrict__ bias,
    float* __restrict__ var)
{
  const int x = threadIdx.x, y = blockIdx.x, b = blockIdx.y;
  __shared__ float sw[289];
  for (int i = threadIdx.x; i < 288; i += 256) sw[i] = w[i];
  if (threadIdx.x == 0) sw[288] = bias[0];
  __syncthreads();
  float acc = sw[288];
  #pragma unroll
  for (int t = 0; t < 9; ++t) {
    int yy = y + t / 3 - 1, xx = x + t % 3 - 1;
    if (yy >= 0 && yy < IMG && xx >= 0 && xx < IMG) {
      const _Float16* src = v2 + (((size_t)b * IMG + yy) * IMG + xx) * 32;
      for (int c = 0; c < 32; ++c) acc = fmaf((float)src[c], sw[c * 9 + t], acc);
    }
  }
  var[((size_t)b * IMG + y) * IMG + x] = acc;
}

// ---------------------------------------------------------------------------
// Affinity weights from NHWC f16 feats (64 ch): contiguous 128B per pixel
// ---------------------------------------------------------------------------
__global__ __launch_bounds__(256) void affinity_edge(
    const _Float16* __restrict__ feats,  // NHWC [b][y][x][64]
    const float* __restrict__ log_mu,
    float* __restrict__ wv, float* __restrict__ wh)
{
  const int x = threadIdx.x, y = blockIdx.x, b = blockIdx.y;
  const float mu = expf(log_mu[0]);
  const _Float16* f0 = feats + (((size_t)b * IMG + y) * IMG + x) * 64;
  float dv = 0.f, dh = 0.f;
  for (int c = 0; c < 64; ++c) {
    float fc = (float)f0[c];
    if (y < IMG - 1) { float d = (float)f0[(size_t)IMG * 64 + c] - fc; dv = fmaf(d, d, dv); }
    if (x < IMG - 1) { float d = (float)f0[64 + c] - fc;               dh = fmaf(d, d, dh); }
  }
  size_t idx = ((size_t)b * IMG + y) * IMG + x;
  wv[idx] = (y < IMG - 1) ? expf(-mu * dv) : 0.f;
  wh[idx] = (x < IMG - 1) ? expf(-mu * dh) : 0.f;
}

__global__ __launch_bounds__(256) void affinity_assemble(
    const float* __restrict__ wv, const float* __restrict__ wh,
    const float* __restrict__ log_lambda, float* __restrict__ aff)
{
  const int x = threadIdx.x, y = blockIdx.x, b = blockIdx.y;
  const float lam = expf(log_lambda[0]);
  size_t idx = ((size_t)b * IMG + y) * IMG + x;
  float up = (y > 0) ? wv[idx - IMG] : 0.f;
  float dn = wv[idx];
  float lf = (x > 0) ? wh[idx - 1] : 0.f;
  float rt = wh[idx];
  float* A = aff + (size_t)b * 5 * HW + (size_t)y * IMG + x;
  A[0] = up; A[HW] = dn; A[2 * HW] = lf; A[3 * HW] = rt;
  A[4 * HW] = up + dn + lf + rt + lam;
}

// ---------------------------------------------------------------------------
// b = lam * up_adj(mask*source, 8)
// ---------------------------------------------------------------------------
__global__ __launch_bounds__(256) void compute_b(
    const float* __restrict__ source, const float* __restrict__ mask,
    const float* __restrict__ log_lambda, float* __restrict__ bvec)
{
  const int x = threadIdx.x, y = blockIdx.x, b = blockIdx.y;
  const float lam = expf(log_lambda[0]);
  int s = (b * 32 + (y >> 3)) * 32 + (x >> 3);
  bvec[((size_t)b * IMG + y) * IMG + x] = lam * mask[s] * source[s] * (1.f / 64.f);
}

// ---------------------------------------------------------------------------
// Deterministic in-block reduction of a 4096-float partial array (L2-hot).
// ---------------------------------------------------------------------------
__device__ __forceinline__ float block_sum4096(const float* __restrict__ part,
                                               float* red, int tid)
{
  float s = 0.f;
  #pragma unroll
  for (int i = 0; i < 16; ++i) s += part[tid + (i << 8)];
  red[tid] = s;
  __syncthreads();
  #pragma unroll
  for (int st = 128; st > 0; st >>= 1) {
    if (tid < st) red[tid] += red[tid + st];
    __syncthreads();
  }
  float t = red[0];
  __syncthreads();
  return t;
}

// ---------------------------------------------------------------------------
// Fused CG step A: p_k = r + beta*p_{k-1} (recomputed incl. halo -> no race),
// Ap = A p_k, partials of <p,Ap>. it==0: p taken directly from pOld (x0/p0).
// ---------------------------------------------------------------------------
__global__ __launch_bounds__(256) void cg_matvec_fused(
    const float* __restrict__ pOld, float* __restrict__ pNew,
    const float* __restrict__ rv,
    const float* __restrict__ partR_cur,  // rs_k partials   (beta numerator)
    const float* __restrict__ partR_old,  // rs_{k-1} partials
    const float* __restrict__ aff, const float* __restrict__ mask,
    const float* __restrict__ log_lambda,
    float* __restrict__ Ap, float* __restrict__ partA, int it)
{
  const int tid = threadIdx.x;
  const int tx = tid & 15, ty = tid >> 4;
  const int x = (blockIdx.x << 4) + tx;
  const int y = (blockIdx.y << 4) + ty;
  const int b = blockIdx.z;
  __shared__ float red[256];

  float beta = 0.f;
  if (it > 0) {
    float rsn = block_sum4096(partR_cur, red, tid);
    float rso = block_sum4096(partR_old, red, tid);
    beta = rsn / rso;
  }
  const float lam = expf(log_lambda[0]);
  const size_t pix = ((size_t)b * IMG + y) * IMG + x;

  auto pval = [&](size_t i) -> float {
    return (it > 0) ? fmaf(beta, pOld[i], rv[i]) : pOld[i];
  };
  const float yc = pval(pix);
  const float yu = (y > 0)       ? pval(pix - IMG) : 0.f;
  const float yd = (y < IMG - 1) ? pval(pix + IMG) : 0.f;
  const float yl = (x > 0)       ? pval(pix - 1)   : 0.f;
  const float yr = (x < IMG - 1) ? pval(pix + 1)   : 0.f;
  if (it > 0) pNew[pix] = yc;

  const float* A = aff + (size_t)b * 5 * HW + (size_t)y * IMG + x;
  const float wu = A[0], wd = A[HW], wl = A[2 * HW], wr = A[3 * HW], ce = A[4 * HW];
  float lap = (ce - lam) * yc - (wu * yu + wd * yd + wl * yl + wr * yr);

  // deterministic 8x8 pooling (4 sub-blocks per 16x16 tile)
  const int sblk = ((ty >> 3) << 1) | (tx >> 3);
  const int sid  = ((ty & 7) << 3) | (tx & 7);
  red[sblk * 64 + sid] = yc;
  __syncthreads();
  #pragma unroll
  for (int s = 32; s > 0; s >>= 1) {
    if (sid < s) red[sblk * 64 + sid] += red[sblk * 64 + sid + s];
    __syncthreads();
  }
  const float psum = red[sblk * 64];
  __syncthreads();
  const int gblk = (b * 32 + (y >> 3)) * 32 + (x >> 3);
  const float ap = lap + lam * mask[gblk] * psum * (1.f / 4096.f);
  Ap[pix] = ap;

  red[tid] = yc * ap;                        // partial <p, Ap>
  __syncthreads();
  #pragma unroll
  for (int s = 128; s > 0; s >>= 1) {
    if (tid < s) red[tid] += red[tid + s];
    __syncthreads();
  }
  if (tid == 0)
    partA[blockIdx.x + 16 * (blockIdx.y + 16 * blockIdx.z)] = red[0];
}

// ---------------------------------------------------------------------------
// Fused CG step B: alpha = rs_k/<p,Ap>; x += alpha p; r -= alpha Ap;
// emits rs_{k+1} partials.
// ---------------------------------------------------------------------------
__global__ __launch_bounds__(256) void cg_update_xr(
    const float* __restrict__ p, const float* __restrict__ Ap,
    float* __restrict__ x, float* __restrict__ r,
    const float* __restrict__ partR_cur, const float* __restrict__ partA,
    float* __restrict__ partR_next)
{
  const int tid = threadIdx.x;
  __shared__ float red[256];
  float rs  = block_sum4096(partR_cur, red, tid);
  float pap = block_sum4096(partA, red, tid);
  float alpha = rs / pap;
  size_t i = (size_t)blockIdx.x * 256 + tid;
  x[i] = fmaf(alpha, p[i], x[i]);
  float rv = fmaf(-alpha, Ap[i], r[i]);
  r[i] = rv;
  red[tid] = rv * rv;
  __syncthreads();
  #pragma unroll
  for (int s = 128; s > 0; s >>= 1) {
    if (tid < s) red[tid] += red[tid + s];
    __syncthreads();
  }
  if (tid == 0) partR_next[blockIdx.x] = red[0];
}

__global__ __launch_bounds__(256) void cg_init(
    const float* __restrict__ bvec, const float* __restrict__ Ax0,
    const float* __restrict__ x0, float* __restrict__ x,
    float* __restrict__ r, float* __restrict__ p, float* __restrict__ partials)
{
  const int tid = threadIdx.x;
  size_t i = (size_t)blockIdx.x * 256 + tid;
  float rvv = bvec[i] - Ax0[i];
  r[i] = rvv; p[i] = rvv; x[i] = x0[i];
  __shared__ float red[256];
  red[tid] = rvv * rvv;
  __syncthreads();
  #pragma unroll
  for (int s = 128; s > 0; s >>= 1) {
    if (tid < s) red[tid] += red[tid + s];
    __syncthreads();
  }
  if (tid == 0) partials[blockIdx.x] = red[0];
}

// ---------------------------------------------------------------------------
extern "C" void kernel_launch(void* const* d_in, const int* in_sizes, int n_in,
                              void* d_out, int out_size, void* d_ws, size_t ws_size,
                              hipStream_t stream)
{
  (void)in_sizes; (void)n_in; (void)out_size; (void)ws_size;

  const float* guide   = (const float*)d_in[0];
  const float* source  = (const float*)d_in[1];
  const float* mask_lr = (const float*)d_in[2];
  const float* y_bic   = (const float*)d_in[3];
  const float* gw1 = (const float*)d_in[4];  const float* gb1 = (const float*)d_in[5];
  const float* gw2 = (const float*)d_in[6];  const float* gb2 = (const float*)d_in[7];
  const float* sw1 = (const float*)d_in[8];  const float* sb1 = (const float*)d_in[9];
  const float* sw2 = (const float*)d_in[10]; const float* sb2 = (const float*)d_in[11];
  const float* vw1 = (const float*)d_in[12]; const float* vb1 = (const float*)d_in[13];
  const float* vw2 = (const float*)d_in[14]; const float* vb2 = (const float*)d_in[15];
  const float* vw3 = (const float*)d_in[16]; const float* vb3 = (const float*)d_in[17];
  const float* log_lambda = (const float*)d_in[18];
  const float* log_mu     = (const float*)d_in[19];

  float* y_pred = (float*)d_out;                   // CG x lives here
  float* var    = y_pred + (size_t)BATCH * HW;
  float* aff    = y_pred + 2 * (size_t)BATCH * HW; // (16,5,256,256)

  char* ws = (char*)d_ws;
  const size_t MB = 1ull << 20;
  _Float16* act   = (_Float16*)(ws + 0);           //  64 MB f16 NHWC layer-1 acts
  _Float16* feats = (_Float16*)(ws + 64 * MB);     // 128 MB f16 NHWC [b][y][x][64]
  _Float16* v2    = feats;                         // v-path reuses (dead before g/s)
  float* wvb   = (float*)(ws + 192 * MB);
  float* whb   = (float*)(ws + 196 * MB);
  float* bvec  = (float*)(ws + 200 * MB);
  float* rvec  = (float*)(ws + 204 * MB);
  float* pb0   = (float*)(ws + 208 * MB);
  float* pb1   = (float*)(ws + 212 * MB);
  float* Apv   = (float*)(ws + 216 * MB);
  _Float16* wpack = (_Float16*)(ws + 220 * MB);    // 18 KB packed B fragments
  float* partR0 = (float*)(ws + 220 * MB + (64 << 10));   // 4096 floats each
  float* partR1 = (float*)(ws + 220 * MB + (80 << 10));
  float* partA  = (float*)(ws + 220 * MB + (96 << 10));

  const dim3 gPix(IMG, BATCH), bPix(256);
  const dim3 gTile(16, 16, 16), bTile(256);

  // ---- variance path
  conv_l1<<<gPix, bPix, 0, stream>>>(guide, 3, y_bic, 1, vw1, vb1, act);
  pack_w<<<36, 256, 0, stream>>>(vw2, wpack);
  conv32_wmma<<<gPix, bPix, 0, stream>>>(act, wpack, vb2, v2, 32, 0, 1);
  conv_vw3<<<gPix, bPix, 0, stream>>>(v2, vw3, vb3, var);

  // ---- feature path (g -> ch 0..31, s -> ch 32..63; no relu on layer 2)
  conv_l1<<<gPix, bPix, 0, stream>>>(guide, 3, (const float*)nullptr, 0, gw1, gb1, act);
  pack_w<<<36, 256, 0, stream>>>(gw2, wpack);
  conv32_wmma<<<gPix, bPix, 0, stream>>>(act, wpack, gb2, feats, 64, 0, 0);

  conv_l1<<<gPix, bPix, 0, stream>>>(y_bic, 1, (const float*)nullptr, 0, sw1, sb1, act);
  pack_w<<<36, 256, 0, stream>>>(sw2, wpack);
  conv32_wmma<<<gPix, bPix, 0, stream>>>(act, wpack, sb2, feats, 64, 32, 0);

  // ---- affinity -> d_out
  affinity_edge<<<gPix, bPix, 0, stream>>>(feats, log_mu, wvb, whb);
  affinity_assemble<<<gPix, bPix, 0, stream>>>(wvb, whb, log_lambda, aff);

  // ---- CG setup
  compute_b<<<gPix, bPix, 0, stream>>>(source, mask_lr, log_lambda, bvec);
  cg_matvec_fused<<<gTile, bTile, 0, stream>>>(y_bic, pb0, rvec, nullptr, nullptr,
                                               aff, mask_lr, log_lambda, Apv, partA, 0);
  cg_init<<<4096, 256, 0, stream>>>(bvec, Apv, y_bic, y_pred, rvec, pb0, partR0);

  // ---- 100 CG iterations, 2 launches each (deterministic reductions)
  for (int k = 0; k < 100; ++k) {
    const float* pOld = (k == 0) ? pb0 : ((k & 1) ? pb0 : pb1);
    float*       pNew = (k & 1) ? pb1 : pb0;
    const float* Rcur = (k & 1) ? partR1 : partR0;   // rs_k partials
    float*       Rnxt = (k & 1) ? partR0 : partR1;   // rs_{k+1} partials
    const float* Rold = (k == 0) ? nullptr : ((k & 1) ? partR0 : partR1);
    cg_matvec_fused<<<gTile, bTile, 0, stream>>>(pOld, pNew, rvec, Rcur, Rold,
                                                 aff, mask_lr, log_lambda,
                                                 Apv, partA, k);
    cg_update_xr<<<4096, 256, 0, stream>>>(pNew, Apv, y_pred, rvec,
                                           Rcur, partA, Rnxt);
  }
}
